// TransitionDown_27436251087201
// MI455X (gfx1250) — matrix operand order repaired and checked by reading, not compiled, not run on hardware
//
#include <hip/hip_runtime.h>
#include <hip/hip_bf16.h>

// ---------------------------------------------------------------------------
// Point Transformer "TransitionDown" for MI455X (gfx1250, wave32, WMMA).
// Pipeline: FPS -> KNN -> gathers/pos-enc-L1 -> bf16 WMMA GEMM chain with
// fused epilogues -> fused softmax+weighted-sum -> final GEMM -> BatchNorm.
// ---------------------------------------------------------------------------

typedef __attribute__((ext_vector_type(8)))  __bf16 v8bf;
typedef __attribute__((ext_vector_type(16))) __bf16 v16bf;
typedef __attribute__((ext_vector_type(8)))  float  v8f;

#define B_   8
#define N_   4096
#define S_   1024
#define K_   16
#define CIN  128
#define COUT 256
#define BS_  (B_ * S_)          // 8192
#define R_   (BS_ * K_)         // 131072

// ---------------------------------------------------------------------------
// 1) Farthest point sampling: one workgroup per batch, xyz in LDS.
// ---------------------------------------------------------------------------
__global__ __launch_bounds__(1024) void fps_kernel(const float* __restrict__ xyz,
                                                   int* __restrict__ fps_idx,
                                                   float* __restrict__ new_xyz) {
  __shared__ float sx[N_ * 3];
  __shared__ float rv[32];
  __shared__ int   ri[32];
  __shared__ int   s_far;
  const int b = blockIdx.x, tid = threadIdx.x;
  const float* xb = xyz + (size_t)b * N_ * 3;
  for (int j = tid; j < N_ * 3; j += 1024) sx[j] = xb[j];
  float dist[4];
#pragma unroll
  for (int i = 0; i < 4; ++i) dist[i] = 1e10f;
  if (tid == 0) s_far = 0;
  __syncthreads();

  for (int s = 0; s < S_; ++s) {
    const int far = s_far;
    if (tid == 0) fps_idx[b * S_ + s] = far;
    if (tid < 3)  new_xyz[(size_t)(b * S_ + s) * 3 + tid] = sx[far * 3 + tid];
    const float cx = sx[far * 3], cy = sx[far * 3 + 1], cz = sx[far * 3 + 2];
    float bv = -1.0f; int bidx = 0;
#pragma unroll
    for (int i = 0; i < 4; ++i) {
      const int p = tid + i * 1024;
      const float dx = sx[p * 3] - cx, dy = sx[p * 3 + 1] - cy, dz = sx[p * 3 + 2] - cz;
      const float d = dx * dx + dy * dy + dz * dz;
      dist[i] = fminf(dist[i], d);
      if (dist[i] > bv) { bv = dist[i]; bidx = p; }
    }
    // wave32 reduce (lower index wins ties -> matches jnp.argmax)
    for (int off = 16; off > 0; off >>= 1) {
      const float ov = __shfl_down(bv, off);
      const int   oi = __shfl_down(bidx, off);
      if (ov > bv || (ov == bv && oi < bidx)) { bv = ov; bidx = oi; }
    }
    if ((tid & 31) == 0) { rv[tid >> 5] = bv; ri[tid >> 5] = bidx; }
    __syncthreads();
    if (tid < 32) {
      bv = rv[tid]; bidx = ri[tid];
      for (int off = 16; off > 0; off >>= 1) {
        const float ov = __shfl_down(bv, off);
        const int   oi = __shfl_down(bidx, off);
        if (ov > bv || (ov == bv && oi < bidx)) { bv = ov; bidx = oi; }
      }
      if (tid == 0) s_far = bidx;
    }
    __syncthreads();
  }
}

// ---------------------------------------------------------------------------
// 2) KNN (top-16 by squared distance), xyz tiled through LDS.
// ---------------------------------------------------------------------------
__global__ __launch_bounds__(256) void knn_kernel(const float* __restrict__ xyz,
                                                  const float* __restrict__ new_xyz,
                                                  int* __restrict__ knn_idx) {
  __shared__ float sx[1024 * 3];
  const int b  = blockIdx.y;
  const int q  = blockIdx.x * 256 + threadIdx.x;
  const int bs = b * S_ + q;
  const float qx = new_xyz[(size_t)bs * 3];
  const float qy = new_xyz[(size_t)bs * 3 + 1];
  const float qz = new_xyz[(size_t)bs * 3 + 2];
  float bd[K_]; int bi[K_];
#pragma unroll
  for (int i = 0; i < K_; ++i) { bd[i] = 3.4e38f; bi[i] = 0; }

  for (int t = 0; t < N_ / 1024; ++t) {
    __syncthreads();
    for (int j = threadIdx.x; j < 1024 * 3; j += 256)
      sx[j] = xyz[((size_t)b * N_ + t * 1024) * 3 + j];
    __syncthreads();
    for (int j = 0; j < 1024; ++j) {
      const float dx = sx[j * 3] - qx, dy = sx[j * 3 + 1] - qy, dz = sx[j * 3 + 2] - qz;
      const float d = dx * dx + dy * dy + dz * dz;
      if (d < bd[K_ - 1]) {
        bd[K_ - 1] = d; bi[K_ - 1] = t * 1024 + j;
#pragma unroll
        for (int p = K_ - 1; p >= 1; --p) {
          if (bd[p] < bd[p - 1]) {
            const float tv = bd[p]; bd[p] = bd[p - 1]; bd[p - 1] = tv;
            const int   ti = bi[p]; bi[p] = bi[p - 1]; bi[p - 1] = ti;
          }
        }
      }
    }
  }
#pragma unroll
  for (int i = 0; i < K_; ++i) knn_idx[(size_t)bs * K_ + i] = bi[i];
}

// ---------------------------------------------------------------------------
// 3) Gathers + positional MLP layer 1.
// ---------------------------------------------------------------------------
__global__ __launch_bounds__(256) void gather_xs(const float* __restrict__ points,
                                                 const int* __restrict__ fps_idx,
                                                 __bf16* __restrict__ xs) {
  const int e = blockIdx.x * 256 + threadIdx.x;       // BS_*CIN threads
  const int bs = e >> 7, c = e & 127, b = bs >> 10;
  xs[e] = (__bf16)points[((size_t)b * N_ + fps_idx[bs]) * CIN + c];
}

__global__ __launch_bounds__(256) void gather_xk(const float* __restrict__ points,
                                                 const float* __restrict__ xyz,
                                                 const float* __restrict__ new_xyz,
                                                 const int* __restrict__ knn_idx,
                                                 __bf16* __restrict__ xk,
                                                 float* __restrict__ xyzn) {
  const int e = blockIdx.x * 256 + threadIdx.x;       // R_*CIN threads
  const int r = e >> 7, c = e & 127;
  const int bs = r >> 4, b = bs >> 10;
  const int idx = knn_idx[r];
  xk[e] = (__bf16)points[((size_t)b * N_ + idx) * CIN + c];
  if (c < 3)
    xyzn[(size_t)r * 3 + c] =
        xyz[((size_t)b * N_ + idx) * 3 + c] - new_xyz[(size_t)bs * 3 + c];
}

// pos1 = relu(xyzn @ dw1 + db1)  (K=3, scalar is optimal) -> bf16
__global__ __launch_bounds__(256) void pos_layer1(const float* __restrict__ xyzn,
                                                  const float* __restrict__ dw1,
                                                  const float* __restrict__ db1,
                                                  __bf16* __restrict__ pos1) {
  const int e = blockIdx.x * 256 + threadIdx.x;       // R_*COUT threads
  const int r = e >> 8, c = e & 255;
  const float x0 = xyzn[(size_t)r * 3], x1 = xyzn[(size_t)r * 3 + 1], x2 = xyzn[(size_t)r * 3 + 2];
  float v = x0 * dw1[c] + x1 * dw1[COUT + c] + x2 * dw1[2 * COUT + c] + db1[c];
  pos1[e] = (__bf16)fmaxf(v, 0.0f);
}

// ---------------------------------------------------------------------------
// 4) Weight pre-pack into the B-fragment layout this GEMM uses (contiguous k per lane half):
//    dst[((kt*16+nt)*32 + lane)*16 + j] =
//        src[(kt*32 + (lane>>4)*16 + j)*256 + nt*16 + (lane&15)]
// ---------------------------------------------------------------------------
__global__ __launch_bounds__(256) void pack_b(const float* __restrict__ src,
                                              __bf16* __restrict__ dst, int Kd) {
  const int e = blockIdx.x * 256 + threadIdx.x;
  if (e >= Kd * COUT) return;
  const int j = e & 15, lane = (e >> 4) & 31, nt = (e >> 9) & 15, kt = e >> 13;
  const int k = kt * 32 + (lane >> 4) * 16 + j;
  const int n = nt * 16 + (lane & 15);
  dst[e] = (__bf16)src[(size_t)k * COUT + n];
}

// ---------------------------------------------------------------------------
// 5) bf16 WMMA GEMM, N fixed = 256. Block = 8 waves, tile 128(M) x 256(N).
//    Wave w: mi=w&3 -> 32-row M subtile (2 A fragments), ni=w>>2 -> 128-col
//    N half (8 B fragments). Each B fragment feeds two WMMAs; pointers
//    advance by constant strides so B loads are one base + immediate
//    offsets and clause together. Unconditional WGP-scope prefetch of the
//    next A tile (speculative, dropped if OOB, no counter traffic).
// ---------------------------------------------------------------------------
template <bool RELU, bool BIAS, bool ADD, bool SUBQ, bool OUTF>
__global__ __launch_bounds__(256) void gemm256_wmma(
    const __bf16* __restrict__ A, const __bf16* __restrict__ Bp, int M, int Kd,
    const float* __restrict__ bias, const __bf16* __restrict__ addv,
    const __bf16* __restrict__ subq, float scale,
    float* __restrict__ outF, __bf16* __restrict__ outH) {
  const int lane = threadIdx.x & 31;
  const int wid  = threadIdx.x >> 5;
  const int mi = wid & 3, ni = wid >> 2;
  const int m0 = blockIdx.x * 128 + mi * 32;
  const int h  = lane >> 4;           // which half of the wave
  const int ml = lane & 15;

  v8f acc[2][8] = {};
  const int KT = Kd >> 5;
  // A: rows m0+ml and m0+16+ml, contiguous K per lane half (matches packed B):
  // lane<16 -> K 0..15, lane>=16 -> K 16..31.
  const __bf16* ap = A + (size_t)(m0 + ml) * Kd + h * 16;
  // B: fragment (kt=0, nt=ni*8), consecutive nt are +512 elements (1024 B).
  const __bf16* bp = Bp + ((size_t)(ni * 8) * 32 + lane) * 16;

  for (int kt = 0; kt < KT; ++kt) {
    v16bf a[2];
#pragma unroll
    for (int mm = 0; mm < 2; ++mm) {
      const __bf16* amp = ap + (size_t)(mm * 16) * Kd;
      const v8bf lo = *(const v8bf*)(amp);
      const v8bf hi = *(const v8bf*)(amp + 8);
#pragma unroll
      for (int j = 0; j < 8; ++j) { a[mm][j] = lo[j]; a[mm][8 + j] = hi[j]; }
    }
    // Speculative near-cache prefetch of next A tile; no branch, no counters.
    __builtin_prefetch(ap + 32, 0, 3);
    v16bf b[8];
#pragma unroll
    for (int t = 0; t < 8; ++t) b[t] = *(const v16bf*)(bp + t * 512);
#pragma unroll
    for (int t = 0; t < 8; ++t) {
#pragma unroll
      for (int mm = 0; mm < 2; ++mm) {
        acc[mm][t] = __builtin_amdgcn_wmma_f32_16x16x32_bf16(
            false, a[mm], false, b[t], (short)0, acc[mm][t], false, false);
      }
    }
    ap += 32;                 // next 32 K-elements of the same rows
    bp += 16 * 32 * 16;       // next kt plane of packed B
  }

  // C/D layout: VGPR i, lane L -> row = (L>>4)*8 + i, col = L&15 (per tile)
#pragma unroll
  for (int mm = 0; mm < 2; ++mm) {
#pragma unroll
    for (int t = 0; t < 8; ++t) {
      const int col = (ni * 8 + t) * 16 + ml;
#pragma unroll
      for (int i = 0; i < 8; ++i) {
        const int r = m0 + mm * 16 + h * 8 + i;
        float x = acc[mm][t][i];
        if (SUBQ) x = (float)subq[((size_t)(r >> 4)) * COUT + col] - x;
        if (ADD)  x += (float)addv[(size_t)r * COUT + col];
        if (BIAS) x += bias[col];
        x *= scale;
        if (RELU) x = fmaxf(x, 0.0f);
        if (OUTF) outF[(size_t)r * COUT + col] = x;
        else      outH[(size_t)r * COUT + col] = (__bf16)x;
      }
    }
  }
}

// ---------------------------------------------------------------------------
// 6) Fused softmax over the K=16 neighbor axis + weighted sum with (v+pos).
// ---------------------------------------------------------------------------
__global__ __launch_bounds__(256) void softmax_reduce(const float* __restrict__ logits,
                                                      const __bf16* __restrict__ vp,
                                                      __bf16* __restrict__ resmid) {
  const int bs = blockIdx.x, c = threadIdx.x;
  float l[K_]; float mx = -3.4e38f;
#pragma unroll
  for (int k = 0; k < K_; ++k) {
    l[k] = logits[((size_t)(bs * K_ + k)) * COUT + c];
    mx = fmaxf(mx, l[k]);
  }
  float sum = 0.0f;
#pragma unroll
  for (int k = 0; k < K_; ++k) { l[k] = __expf(l[k] - mx); sum += l[k]; }
  const float inv = 1.0f / sum;
  float acc = 0.0f;
#pragma unroll
  for (int k = 0; k < K_; ++k)
    acc += l[k] * inv * (float)vp[((size_t)(bs * K_ + k)) * COUT + c];
  resmid[(size_t)bs * COUT + c] = (__bf16)acc;
}

// ---------------------------------------------------------------------------
// 7) BatchNorm (training-mode batch stats) + ReLU
// ---------------------------------------------------------------------------
__global__ void zero_stats(float* __restrict__ stats) {
  const int i = threadIdx.x;
  stats[i] = 0.0f; stats[COUT + i] = 0.0f;
}

__global__ __launch_bounds__(256) void bn_reduce(const float* __restrict__ res2,
                                                 float* __restrict__ stats) {
  const int c = threadIdx.x;
  const int r0 = blockIdx.x * 256;
  float s = 0.0f, ss = 0.0f;
  for (int r = r0; r < r0 + 256; ++r) {
    const float x = res2[(size_t)r * COUT + c];
    s += x; ss += x * x;
  }
  atomicAdd(&stats[c], s);
  atomicAdd(&stats[COUT + c], ss);
}

__global__ __launch_bounds__(256) void bn_finalize(const float* __restrict__ res2,
                                                   const float* __restrict__ stats,
                                                   const float* __restrict__ g,
                                                   const float* __restrict__ bta,
                                                   float* __restrict__ out) {
  const int c = threadIdx.x, r = blockIdx.x;
  const float mean = stats[c] * (1.0f / (float)BS_);
  const float var  = stats[COUT + c] * (1.0f / (float)BS_) - mean * mean;
  const float x = res2[(size_t)r * COUT + c];
  const float y = g[c] * (x - mean) * rsqrtf(var + 1e-5f) + bta[c];
  out[(size_t)r * COUT + c] = fmaxf(y, 0.0f);
}

// ---------------------------------------------------------------------------
// Host launcher
// ---------------------------------------------------------------------------
static inline size_t align256(size_t x) { return (x + 255) & ~(size_t)255; }

extern "C" void kernel_launch(void* const* d_in, const int* in_sizes, int n_in,
                              void* d_out, int out_size, void* d_ws, size_t ws_size,
                              hipStream_t stream) {
  (void)in_sizes; (void)n_in; (void)out_size; (void)ws_size;
  const float* xyz    = (const float*)d_in[0];
  const float* points = (const float*)d_in[1];
  const float* wq  = (const float*)d_in[2];
  const float* wk  = (const float*)d_in[3];
  const float* wv  = (const float*)d_in[4];
  const float* dw1 = (const float*)d_in[5];
  const float* db1 = (const float*)d_in[6];
  const float* dw2 = (const float*)d_in[7];
  const float* db2 = (const float*)d_in[8];
  const float* gw1 = (const float*)d_in[9];
  const float* gb1 = (const float*)d_in[10];
  const float* gw2 = (const float*)d_in[11];
  const float* gb2 = (const float*)d_in[12];
  const float* lw  = (const float*)d_in[13];
  const float* lb  = (const float*)d_in[14];
  const float* bng = (const float*)d_in[15];
  const float* bnb = (const float*)d_in[16];

  float* out_xyz = (float*)d_out;                 // [B,S,3]
  float* out_res = (float*)d_out + B_ * S_ * 3;   // [B,S,COUT]

  // workspace carve
  char* ws = (char*)d_ws;
  size_t off = 0;
  auto carve = [&](size_t bytes) { void* p = ws + off; off += align256(bytes); return p; };
  int*    fps_idx = (int*)   carve((size_t)BS_ * 4);
  int*    knn_idx = (int*)   carve((size_t)R_ * 4);
  float*  xyzn    = (float*) carve((size_t)R_ * 3 * 4);
  __bf16* xs      = (__bf16*)carve((size_t)BS_ * CIN * 2);
  __bf16* xk      = (__bf16*)carve((size_t)R_ * CIN * 2);
  __bf16* qb      = (__bf16*)carve((size_t)BS_ * COUT * 2);
  __bf16* pos1    = (__bf16*)carve((size_t)R_ * COUT * 2);   // reused as h
  __bf16* pos     = (__bf16*)carve((size_t)R_ * COUT * 2);
  __bf16* tbuf    = (__bf16*)carve((size_t)R_ * COUT * 2);
  __bf16* vp      = (__bf16*)carve((size_t)R_ * COUT * 2);
  float*  logits  = (float*) carve((size_t)R_ * COUT * 4);
  __bf16* resmid  = (__bf16*)carve((size_t)BS_ * COUT * 2);
  float*  res2    = (float*) carve((size_t)BS_ * COUT * 4);
  float*  stats   = (float*) carve(2 * COUT * 4);
  __bf16* wqP  = (__bf16*)carve((size_t)CIN * COUT * 2);
  __bf16* wkP  = (__bf16*)carve((size_t)CIN * COUT * 2);
  __bf16* wvP  = (__bf16*)carve((size_t)CIN * COUT * 2);
  __bf16* dw2P = (__bf16*)carve((size_t)COUT * COUT * 2);
  __bf16* gw1P = (__bf16*)carve((size_t)COUT * COUT * 2);
  __bf16* gw2P = (__bf16*)carve((size_t)COUT * COUT * 2);
  __bf16* lwP  = (__bf16*)carve((size_t)COUT * COUT * 2);
  __bf16* hbuf = pos1;  // alias: pos1 dead after pos GEMM

  // --- sampling & grouping ---
  fps_kernel<<<B_, 1024, 0, stream>>>(xyz, fps_idx, out_xyz);
  knn_kernel<<<dim3(S_ / 256, B_), 256, 0, stream>>>(xyz, out_xyz, knn_idx);
  gather_xs<<<(BS_ * CIN) / 256, 256, 0, stream>>>(points, fps_idx, xs);
  gather_xk<<<(R_ * CIN) / 256, 256, 0, stream>>>(points, xyz, out_xyz, knn_idx, xk, xyzn);
  pos_layer1<<<(R_ * COUT) / 256, 256, 0, stream>>>(xyzn, dw1, db1, pos1);

  // --- weight packing into WMMA B-fragment layout ---
  pack_b<<<(CIN * COUT) / 256, 256, 0, stream>>>(wq, wqP, CIN);
  pack_b<<<(CIN * COUT) / 256, 256, 0, stream>>>(wk, wkP, CIN);
  pack_b<<<(CIN * COUT) / 256, 256, 0, stream>>>(wv, wvP, CIN);
  pack_b<<<(COUT * COUT) / 256, 256, 0, stream>>>(dw2, dw2P, COUT);
  pack_b<<<(COUT * COUT) / 256, 256, 0, stream>>>(gw1, gw1P, COUT);
  pack_b<<<(COUT * COUT) / 256, 256, 0, stream>>>(gw2, gw2P, COUT);
  pack_b<<<(COUT * COUT) / 256, 256, 0, stream>>>(lw, lwP, COUT);

  // --- WMMA GEMM chain (fused epilogues), block tile 128x256 ---
  // q = xs @ wq -> bf16
  gemm256_wmma<false, false, false, false, false><<<BS_ / 128, 256, 0, stream>>>(
      xs, wqP, BS_, CIN, nullptr, nullptr, nullptr, 1.0f, nullptr, qb);
  // pos = pos1 @ dw2 + db2 -> bf16
  gemm256_wmma<false, true, false, false, false><<<R_ / 128, 256, 0, stream>>>(
      pos1, dw2P, R_, COUT, db2, nullptr, nullptr, 1.0f, nullptr, pos);
  // t = q - (xk @ wk) + pos -> bf16
  gemm256_wmma<false, false, true, true, false><<<R_ / 128, 256, 0, stream>>>(
      xk, wkP, R_, CIN, nullptr, pos, qb, 1.0f, nullptr, tbuf);
  // vp = (xk @ wv) + pos -> bf16
  gemm256_wmma<false, false, true, false, false><<<R_ / 128, 256, 0, stream>>>(
      xk, wvP, R_, CIN, nullptr, pos, nullptr, 1.0f, nullptr, vp);
  // h = relu(t @ gw1 + gb1) -> bf16 (into pos1 buffer)
  gemm256_wmma<true, true, false, false, false><<<R_ / 128, 256, 0, stream>>>(
      tbuf, gw1P, R_, COUT, gb1, nullptr, nullptr, 1.0f, nullptr, hbuf);
  // logits = (h @ gw2 + gb2) / sqrt(COUT) -> f32
  gemm256_wmma<false, true, false, false, true><<<R_ / 128, 256, 0, stream>>>(
      hbuf, gw2P, R_, COUT, gb2, nullptr, nullptr, 0.0625f, logits, nullptr);

  // --- softmax over neighbors + weighted sum ---
  softmax_reduce<<<BS_, 256, 0, stream>>>(logits, vp, resmid);

  // res2 = resmid @ lw + lb -> f32
  gemm256_wmma<false, true, false, false, true><<<BS_ / 128, 256, 0, stream>>>(
      resmid, lwP, BS_, COUT, lb, nullptr, nullptr, 1.0f, res2, nullptr);

  // --- BatchNorm1d (batch stats) + ReLU ---
  zero_stats<<<1, COUT, 0, stream>>>(stats);
  bn_reduce<<<BS_ / 256, 256, 0, stream>>>(res2, stats);
  bn_finalize<<<BS_, 256, 0, stream>>>(res2, stats, bng, bnb, out_res);
}
